// TransformerBlock_627065225346
// MI455X (gfx1250) — matrix operand order repaired and checked by reading, not compiled, not run on hardware
//
#include <hip/hip_runtime.h>
#include <hip/hip_bf16.h>
#include <math.h>

// TransformerBlock for MI455X (gfx1250).
// GEMMs + attention on v_wmma_f32_16x16x32_bf16 (fp32 accumulate).
// GEMM tiles are moved by the Tensor Data Mover (tensor_load_to_lds,
// TENSORcnt-gated, double-buffered LDS) instead of VGPR staging.
//
// B=4 S=1024 D=1024 H=16 DH=64 F=4096.  Workspace: ~216 MB.

#define B_ 4
#define S_ 1024
#define D_ 1024
#define H_ 16
#define DH_ 64
#define F_ 4096
#define EPS_ 1e-6f

typedef __attribute__((ext_vector_type(16))) __bf16 v16bf;
typedef __attribute__((ext_vector_type(8)))  float  v8f;
typedef __attribute__((ext_vector_type(4)))  unsigned v4u;
typedef __attribute__((ext_vector_type(8)))  int    v8i;
typedef __attribute__((ext_vector_type(4)))  int    v4i;

__device__ __forceinline__ unsigned short f2b(float f) {
  unsigned u = __builtin_bit_cast(unsigned, f);
  u += 0x7FFFu + ((u >> 16) & 1u);          // round-to-nearest-even
  return (unsigned short)(u >> 16);
}
__device__ __forceinline__ __bf16 b2h(unsigned short b) {
  return __builtin_bit_cast(__bf16, b);
}

// 16 contiguous bf16 bits -> v16bf  (B-fragment: lane=column, elems=K)
__device__ __forceinline__ v16bf ld16(const unsigned short* p) {
  v16bf r;
#pragma unroll
  for (int i = 0; i < 16; ++i) r[i] = b2h(p[i]);
  return r;
}

// A-fragment gather over a 32-wide K window.
// A layout (matches B fragments): half 0 -> K 0..15 ; half 1 -> K 16..31
__device__ __forceinline__ v16bf ldafrag(const unsigned short* p, int hi) {
  v16bf r;
#pragma unroll
  for (int i = 0; i < 8; ++i) r[i]     = b2h(p[hi * 16 + i]);
#pragma unroll
  for (int i = 0; i < 8; ++i) r[8 + i] = b2h(p[hi * 16 + 8 + i]);
  return r;
}

// ------------------------------------------------------------------ TDM ----
// 2-D tile DMA: 128 rows x 32 bf16 elements, row stride = strideElems,
// into LDS at byte offset lds_off with a 16-byte pad per 64-byte row
// (=> padded LDS row of 40 u16, matching the fragment loaders).
// D# layout per CDNA5 ISA.
__device__ __forceinline__ void tdm_tile_load(const unsigned short* gsrc,
                                              unsigned lds_off, int strideElems) {
  unsigned long long ga = (unsigned long long)(const void*)gsrc;
  v4u g0;
  g0[0] = 1u;                                   // count=1 (user descriptor)
  g0[1] = lds_off;                              // lds_addr (bytes)
  g0[2] = (unsigned)ga;                         // global_addr[95:64]
  g0[3] = (unsigned)(ga >> 32) | (2u << 30);    // addr hi bits + type=2 (image)
  v8i g1;
  // data_size=1 (2B) | pad_enable | pad_interval=3 (16 DW) | pad_amount=3 (4 DW)
  g1[0] = (int)((1u << 16) | (1u << 20) | (3u << 22) | (3u << 25));
  g1[1] = (int)(32u << 16);                     // tensor_dim0 = 32 (tile box)
  g1[2] = (int)(128u << 16);                    // tensor_dim1 = 128
  g1[3] = (int)(32u << 16);                     // tile_dim0 = 32
  g1[4] = 128;                                  // tile_dim1 = 128, tile_dim2 = 0
  g1[5] = strideElems;                          // tensor_dim0_stride (lo32)
  g1[6] = 0;
  g1[7] = 0;
  v4i z4 = {0, 0, 0, 0};
#if defined(__clang_major__) && (__clang_major__ >= 23)
  v8i z8 = {0, 0, 0, 0, 0, 0, 0, 0};
  __builtin_amdgcn_tensor_load_to_lds(g0, g1, z4, z4, z8, 0);
#else
  __builtin_amdgcn_tensor_load_to_lds(g0, g1, z4, z4, 0);
#endif
}

// ---------------------------------------------------------------- GEMM ----
// C[M,N] = A(bf16,[M,K]) @ WT(bf16,[N,K])^T + bias, epilogue per MODE.
// TDM double-buffered tiles; 8 waves in 2x4 grid, wave tile 64x32.
enum { MODE_F32 = 0, MODE_BF16 = 1, MODE_RES = 2, MODE_GEGLU = 3 };

template <int MODE>
__global__ __launch_bounds__(256)
void gemm_tdm(const unsigned short* __restrict__ A,
              const unsigned short* __restrict__ WT,
              const float* __restrict__ bias, const float* __restrict__ resaux,
              float* __restrict__ outF, unsigned short* __restrict__ outB,
              int M, int N, int K) {
  extern __shared__ unsigned short smem[];      // 2 x (A-tile + B-tile), 40 KB
  const unsigned TILE_E = 128 * 40;             // u16 per padded tile (10240 B)
  const unsigned TILE_B = TILE_E * 2;           // bytes
  const int tid = threadIdx.x;
  const int w = tid >> 5, lane = tid & 31;
  const int r = lane & 15, hi = lane >> 4;
  const int wm = w & 1, wn = w >> 1;
  const int m0 = blockIdx.y * 128;
  const int n0 = blockIdx.x * 128;

  v8f acc[4][2] = {};

  const int nk = K >> 5;
  if (w == 0) {  // wave 0 drives the TDM
    tdm_tile_load(A + (size_t)m0 * K, 0u, K);
    tdm_tile_load(WT + (size_t)n0 * K, TILE_B, K);
  }
  for (int it = 0; it < nk; ++it) {
    const int buf = it & 1;
    const unsigned short* As = smem + (unsigned)buf * TILE_E * 2u;
    const unsigned short* Bs = As + TILE_E;
    if (w == 0) {
      if (it + 1 < nk) {  // prefetch next tiles into the other buffer
        const unsigned nb = (unsigned)(buf ^ 1) * TILE_B * 2u;
        const int kb = (it + 1) << 5;
        tdm_tile_load(A + (size_t)m0 * K + kb, nb, K);
        tdm_tile_load(WT + (size_t)n0 * K + kb, nb + TILE_B, K);
        __builtin_amdgcn_s_wait_tensorcnt(2);   // current 2 tiles landed
      } else {
        __builtin_amdgcn_s_wait_tensorcnt(0);
      }
    }
    __syncthreads();                            // publish tiles to all waves

    v16bf af[4];
#pragma unroll
    for (int mt = 0; mt < 4; ++mt)
      af[mt] = ldafrag(&As[(unsigned)(wm * 64 + mt * 16 + r) * 40u], hi);
    v16bf bfr[2];
#pragma unroll
    for (int nt = 0; nt < 2; ++nt)
      bfr[nt] = ld16(&Bs[(unsigned)(wn * 32 + nt * 16 + r) * 40u + hi * 16]);
#pragma unroll
    for (int mt = 0; mt < 4; ++mt)
#pragma unroll
      for (int nt = 0; nt < 2; ++nt)
        acc[mt][nt] = __builtin_amdgcn_wmma_f32_16x16x32_bf16(
            false, af[mt], false, bfr[nt], (short)0, acc[mt][nt], false, false);
    __syncthreads();                            // reads done before overwrite
  }

  // C layout: m_local = j + 8*hi, n_local = r
#pragma unroll
  for (int mt = 0; mt < 4; ++mt)
#pragma unroll
    for (int nt = 0; nt < 2; ++nt)
#pragma unroll
      for (int j = 0; j < 8; ++j) {
        int row = m0 + wm * 64 + mt * 16 + j + 8 * hi;
        int col = n0 + wn * 32 + nt * 16 + r;
        size_t idx = (size_t)row * N + col;
        float v = acc[mt][nt][j] + bias[col];
        if (MODE == MODE_F32) {
          outF[idx] = v;
        } else if (MODE == MODE_BF16) {
          outB[idx] = f2b(v);
        } else if (MODE == MODE_RES) {
          outF[idx] = v + resaux[idx];
        } else { // GEGLU: out = x1 * gelu_tanh(x2)
          float g = 0.5f * v *
                    (1.f + tanhf(0.7978845608028654f * (v + 0.044715f * v * v * v)));
          outB[idx] = f2b(resaux[idx] * g);
        }
      }
}

// --------------------------------------------- weight transpose+convert ----
// WT[n][k] = bf16(W[k][n]); 32x32 LDS tile per 256-thread block.
__global__ __launch_bounds__(256)
void wconv_t(const float* __restrict__ W, unsigned short* __restrict__ WT,
             int K, int N) {
  __shared__ unsigned short t[32][33];
  const int n0 = blockIdx.x * 32, k0 = blockIdx.y * 32;
  const int tx = threadIdx.x & 31, ty = threadIdx.x >> 5;
#pragma unroll
  for (int j = 0; j < 4; ++j) {
    int k = ty + j * 8;
    t[k][tx] = f2b(W[(size_t)(k0 + k) * N + n0 + tx]);
  }
  __syncthreads();
#pragma unroll
  for (int j = 0; j < 4; ++j) {
    int n = ty + j * 8;
    WT[(size_t)(n0 + n) * K + k0 + tx] = t[tx][n];
  }
}

// ------------------------------------------------------------ LayerNorm ----
__global__ __launch_bounds__(256)
void ln_rows(const float* __restrict__ X, const float* __restrict__ sc,
             const float* __restrict__ bi, unsigned short* __restrict__ Y) {
  const int row = blockIdx.x;
  const int tid = threadIdx.x;
  const float* xr = X + (size_t)row * D_;
  __shared__ float ssum[8], ssq[8], smean, srstd;
  float v[4];
  float s = 0.f;
#pragma unroll
  for (int j = 0; j < 4; ++j) { v[j] = xr[tid + j * 256]; s += v[j]; }
#pragma unroll
  for (int m = 1; m < 32; m <<= 1) s += __shfl_xor(s, m, 32);
  if ((tid & 31) == 0) ssum[tid >> 5] = s;
  __syncthreads();
  if (tid == 0) {
    float t = 0.f;
    for (int i = 0; i < 8; ++i) t += ssum[i];
    smean = t / (float)D_;
  }
  __syncthreads();
  const float mean = smean;
  float q = 0.f;
#pragma unroll
  for (int j = 0; j < 4; ++j) { float d = v[j] - mean; q += d * d; }
#pragma unroll
  for (int m = 1; m < 32; m <<= 1) q += __shfl_xor(q, m, 32);
  if ((tid & 31) == 0) ssq[tid >> 5] = q;
  __syncthreads();
  if (tid == 0) {
    float t = 0.f;
    for (int i = 0; i < 8; ++i) t += ssq[i];
    srstd = rsqrtf(t / (float)D_ + EPS_);
  }
  __syncthreads();
  const float rstd = srstd;
  unsigned short* yr = Y + (size_t)row * D_;
#pragma unroll
  for (int j = 0; j < 4; ++j) {
    int col = tid + j * 256;
    yr[col] = f2b((v[j] - mean) * rstd * sc[col] + bi[col]);
  }
}

// ------------------------------------------------------------- QK-norm ----
__global__ __launch_bounds__(256)
void qk_norm(const float* __restrict__ X, const float* __restrict__ sc,
             const float* __restrict__ bi, unsigned short* __restrict__ Y,
             int nrows) {
  const int w = threadIdx.x >> 5, lane = threadIdx.x & 31;
  const int row = blockIdx.x * 8 + w;
  if (row >= nrows) return;
  const float* xr = X + (size_t)row * DH_;
  float a = xr[lane * 2], b = xr[lane * 2 + 1];
  float s = a + b;
#pragma unroll
  for (int m = 1; m < 32; m <<= 1) s += __shfl_xor(s, m, 32);
  const float mean = s * (1.f / DH_);
  float d0 = a - mean, d1 = b - mean;
  float q = d0 * d0 + d1 * d1;
#pragma unroll
  for (int m = 1; m < 32; m <<= 1) q += __shfl_xor(q, m, 32);
  const float rstd = rsqrtf(q * (1.f / DH_) + EPS_);
  const int h = row & (H_ - 1);
  const int c0 = h * DH_ + lane * 2;
  unsigned short* yr = Y + (size_t)row * DH_;
  yr[lane * 2]     = f2b(d0 * rstd * sc[c0] + bi[c0]);
  yr[lane * 2 + 1] = f2b(d1 * rstd * sc[c0 + 1] + bi[c0 + 1]);
}

// ------------------------------------------------------- Flash attention ----
__global__ __launch_bounds__(256)
void flash_attn(const unsigned short* __restrict__ Q,
                const unsigned short* __restrict__ Kx,
                const unsigned short* __restrict__ Vx,
                unsigned short* __restrict__ O) {
  __shared__ unsigned short Ks[32][DH_ + 8];   // [key][dh]
  __shared__ unsigned short VsT[DH_][32 + 8];  // [dh][key]
  __shared__ unsigned short Ps[8][16][32 + 8]; // per-wave P tile (C->A relayout)
  const int tid = threadIdx.x;
  const int w = tid >> 5, lane = tid & 31;
  const int r = lane & 15, hi = lane >> 4;
  const int bh = blockIdx.y;
  const int b = bh >> 4, h = bh & 15;
  const int q0 = blockIdx.x * 128;
  const int qrow = q0 + w * 16 + r;

  const unsigned short* qp = Q + (((size_t)b * S_ + qrow) * H_ + h) * DH_;
  v16bf qf[2];
#pragma unroll
  for (int kk = 0; kk < 2; ++kk) qf[kk] = ldafrag(qp + kk * 32, hi);

  float mrow[8], lrow[8];
  v8f of[4] = {};
#pragma unroll
  for (int j = 0; j < 8; ++j) { mrow[j] = -3.0e38f; lrow[j] = 0.f; }

  for (int kt = 0; kt < S_; kt += 32) {
    { // cooperative K/V tile load (K row-major, V transposed)
      int key = tid >> 3, part = tid & 7;
      size_t base = (((size_t)b * S_ + kt + key) * H_ + h) * DH_ + part * 8;
      const uint4* k4 = reinterpret_cast<const uint4*>(Kx + base);
      uint4* d4 = reinterpret_cast<uint4*>(&Ks[key][part * 8]);
      d4[0] = k4[0];
      const unsigned short* vsrc = Vx + base;
#pragma unroll
      for (int i = 0; i < 8; ++i) VsT[part * 8 + i][key] = vsrc[i];
    }
    __syncthreads();

    v8f s0 = {}, s1 = {};
#pragma unroll
    for (int kk = 0; kk < 2; ++kk) {
      v16bf bk0 = ld16(&Ks[r][kk * 32 + hi * 16]);
      v16bf bk1 = ld16(&Ks[16 + r][kk * 32 + hi * 16]);
      s0 = __builtin_amdgcn_wmma_f32_16x16x32_bf16(false, qf[kk], false, bk0,
                                                   (short)0, s0, false, false);
      s1 = __builtin_amdgcn_wmma_f32_16x16x32_bf16(false, qf[kk], false, bk1,
                                                   (short)0, s1, false, false);
    }

    // online softmax: rows live in 16-lane halves -> xor masks 1..8
#pragma unroll
    for (int j = 0; j < 8; ++j) {
      float a = s0[j] * 0.125f, c = s1[j] * 0.125f;
      float mx = fmaxf(a, c);
#pragma unroll
      for (int m = 1; m < 16; m <<= 1) mx = fmaxf(mx, __shfl_xor(mx, m, 32));
      float mnew = fmaxf(mrow[j], mx);
      float alpha = __expf(mrow[j] - mnew);
      mrow[j] = mnew;
      float p0 = __expf(a - mnew), p1 = __expf(c - mnew);
      float rs = p0 + p1;
#pragma unroll
      for (int m = 1; m < 16; m <<= 1) rs += __shfl_xor(rs, m, 32);
      lrow[j] = lrow[j] * alpha + rs;
#pragma unroll
      for (int t = 0; t < 4; ++t) of[t][j] *= alpha;
      int mloc = j + 8 * hi;
      Ps[w][mloc][r]      = f2b(p0);
      Ps[w][mloc][16 + r] = f2b(p1);
    }

    v16bf pf = ldafrag(&Ps[w][r][0], hi);
#pragma unroll
    for (int t = 0; t < 4; ++t) {
      v16bf bv = ld16(&VsT[t * 16 + r][hi * 16]);
      of[t] = __builtin_amdgcn_wmma_f32_16x16x32_bf16(false, pf, false, bv,
                                                      (short)0, of[t], false, false);
    }
    __syncthreads();
  }

#pragma unroll
  for (int t = 0; t < 4; ++t)
#pragma unroll
    for (int j = 0; j < 8; ++j) {
      int srow = q0 + w * 16 + j + 8 * hi;
      int col = h * DH_ + t * 16 + r;
      O[((size_t)b * S_ + srow) * D_ + col] = f2b(of[t][j] / lrow[j]);
    }
}

// ---------------------------------------------------------------- driver ----
extern "C" void kernel_launch(void* const* d_in, const int* in_sizes, int n_in,
                              void* d_out, int out_size, void* d_ws, size_t ws_size,
                              hipStream_t stream) {
  (void)in_sizes; (void)n_in; (void)out_size; (void)ws_size;
  const float* x    = (const float*)d_in[0];
  /* d_in[1] = z, unused by reference */
  const float* ln1s = (const float*)d_in[2];
  const float* ln1b = (const float*)d_in[3];
  const float* Wq   = (const float*)d_in[4];
  const float* bq   = (const float*)d_in[5];
  const float* Wk   = (const float*)d_in[6];
  const float* bk   = (const float*)d_in[7];
  const float* Wv   = (const float*)d_in[8];
  const float* bv   = (const float*)d_in[9];
  const float* qns  = (const float*)d_in[10];
  const float* qnb  = (const float*)d_in[11];
  const float* kns  = (const float*)d_in[12];
  const float* knb  = (const float*)d_in[13];
  const float* Wo   = (const float*)d_in[14];
  const float* bo   = (const float*)d_in[15];
  const float* ln2s = (const float*)d_in[16];
  const float* ln2b = (const float*)d_in[17];
  const float* W1   = (const float*)d_in[18];
  const float* b1   = (const float*)d_in[19];
  const float* W2   = (const float*)d_in[20];
  const float* b2   = (const float*)d_in[21];
  const float* W3   = (const float*)d_in[22];
  const float* b3   = (const float*)d_in[23];
  float* out = (float*)d_out;

  char* ws = (char*)d_ws;
  const size_t MB = 1ull << 20;
  unsigned short* xn  = (unsigned short*)(ws + 0);         //  8 MB normalized (bf16)
  float*          qf  = (float*)(ws + 8 * MB);             // 16 MB q fp32
  float*          kf  = (float*)(ws + 24 * MB);            // 16 MB k fp32
  unsigned short* qb  = (unsigned short*)(ws + 40 * MB);   //  8 MB q bf16 (normed)
  unsigned short* kb_ = (unsigned short*)(ws + 48 * MB);   //  8 MB k bf16 (normed)
  unsigned short* vb_ = (unsigned short*)(ws + 56 * MB);   //  8 MB v bf16
  unsigned short* ao  = (unsigned short*)(ws + 64 * MB);   //  8 MB attention out bf16
  float*          x2r = (float*)(ws + 72 * MB);            // 16 MB post-attn residual
  float*          x1  = (float*)(ws + 88 * MB);            // 64 MB x1 fp32
  unsigned short* hb  = (unsigned short*)(ws + 152 * MB);  // 32 MB GeGLU out bf16
  unsigned short* wtq = (unsigned short*)(ws + 184 * MB);  //  2 MB Wq^T bf16
  unsigned short* wtk = (unsigned short*)(ws + 186 * MB);
  unsigned short* wtv = (unsigned short*)(ws + 188 * MB);
  unsigned short* wto = (unsigned short*)(ws + 190 * MB);
  unsigned short* wt1 = (unsigned short*)(ws + 192 * MB);  //  8 MB W1^T
  unsigned short* wt2 = (unsigned short*)(ws + 200 * MB);  //  8 MB W2^T
  unsigned short* wt3 = (unsigned short*)(ws + 208 * MB);  //  8 MB W3^T (ends 216 MB)

  const int M = B_ * S_;
  dim3 blk(256);
  dim3 gD(D_ / 128, M / 128);
  dim3 gF(F_ / 128, M / 128);
  const unsigned LDSB = 4u * 128u * 40u * 2u;  // 40 KB double-buffered tiles

  // weight transpose+convert pre-passes (bf16, K-contiguous rows for the TDM)
  dim3 gtDD(D_ / 32, D_ / 32), gtDF(F_ / 32, D_ / 32), gtFD(D_ / 32, F_ / 32);
  wconv_t<<<gtDD, blk, 0, stream>>>(Wq, wtq, D_, D_);
  wconv_t<<<gtDD, blk, 0, stream>>>(Wk, wtk, D_, D_);
  wconv_t<<<gtDD, blk, 0, stream>>>(Wv, wtv, D_, D_);
  wconv_t<<<gtDD, blk, 0, stream>>>(Wo, wto, D_, D_);
  wconv_t<<<gtDF, blk, 0, stream>>>(W1, wt1, D_, F_);
  wconv_t<<<gtDF, blk, 0, stream>>>(W2, wt2, D_, F_);
  wconv_t<<<gtFD, blk, 0, stream>>>(W3, wt3, F_, D_);

  ln_rows<<<dim3(M), blk, 0, stream>>>(x, ln1s, ln1b, xn);

  gemm_tdm<MODE_F32><<<gD, blk, LDSB, stream>>>(xn, wtq, bq, nullptr, qf, nullptr, M, D_, D_);
  gemm_tdm<MODE_F32><<<gD, blk, LDSB, stream>>>(xn, wtk, bk, nullptr, kf, nullptr, M, D_, D_);
  gemm_tdm<MODE_BF16><<<gD, blk, LDSB, stream>>>(xn, wtv, bv, nullptr, nullptr, vb_, M, D_, D_);

  const int nrows = M * H_;
  qk_norm<<<dim3(nrows / 8), blk, 0, stream>>>(qf, qns, qnb, qb, nrows);
  qk_norm<<<dim3(nrows / 8), blk, 0, stream>>>(kf, kns, knb, kb_, nrows);

  flash_attn<<<dim3(S_ / 128, B_ * H_), blk, 0, stream>>>(qb, kb_, vb_, ao);

  gemm_tdm<MODE_RES><<<gD, blk, LDSB, stream>>>(ao, wto, bo, x, x2r, nullptr, M, D_, D_);

  ln_rows<<<dim3(M), blk, 0, stream>>>(x2r, ln2s, ln2b, xn);

  gemm_tdm<MODE_F32><<<gF, blk, LDSB, stream>>>(xn, wt1, b1, nullptr, x1, nullptr, M, F_, D_);
  gemm_tdm<MODE_GEGLU><<<gF, blk, LDSB, stream>>>(xn, wt2, b2, x1, nullptr, hb, M, F_, D_);

  gemm_tdm<MODE_RES><<<gD, blk, LDSB, stream>>>(hb, wt3, b3, x2r, out, nullptr, M, D_, F_);
}
